// BipartiteHeteroGNN_89550068121875
// MI455X (gfx1250) — matrix operand order repaired and checked by reading, not compiled, not run on hardware
//
#include <hip/hip_runtime.h>
#include <hip/hip_bf16.h>

typedef __attribute__((ext_vector_type(16))) __bf16 v16bf;
typedef __attribute__((ext_vector_type(8)))  float  v8f;

#define NV 100000
#define NC 50000
#define NEDGE 1000000
#define GEPS 1e-7f

// ---------------------------------------------------------------------------
// Pack f32 weight W[K][N] (row major) into WMMA bf16 B-fragments.
// Tile (kt,nt) covers K rows [kt*32,kt*32+32), N cols [nt*16,nt*16+16).
// Lane l: column n = nt*16 + (l&15); half h = l>>4; element j holds
// B[kt*32 + 16h + j][n]  (j = 0..15, 2 bf16 per VGPR, low K in low bits).
// ---------------------------------------------------------------------------
__global__ void pack_w(const float* __restrict__ W, __bf16* __restrict__ dst,
                       int K, int N) {
  int idx = blockIdx.x * 256 + threadIdx.x;
  if (idx >= K * N) return;
  int j      = idx & 15;
  int lane   = (idx >> 4) & 31;
  int t      = idx >> 9;              // tile = kt*ntiles + nt
  int ntiles = N >> 4;
  int nt = t % ntiles;
  int kt = t / ntiles;
  int n = nt * 16 + (lane & 15);
  int k = kt * 32 + ((lane >> 4) << 4) + j;
  dst[idx] = (__bf16)W[(size_t)k * N + n];
}

// ---------------------------------------------------------------------------
// C[n,N] = act( A[n,K] @ Bpacked + bias (+ addsrc) ), f32 in/out, bf16 WMMA.
// One wave owns one 16-row strip and ALL 8 n-tiles of its 128-col slice:
// A fragment loaded once per K-step, then 8 independent WMMAs (distinct
// accumulators -> no RAW chains, keeps the matrix pipe saturated).
// A layout: lane m=l&15 is row, half h=l>>4, K runs [16h..16h+15].
// D layout: lane l, vgpr v -> row = 8*(l>>4)+v, col = l&15.
// ---------------------------------------------------------------------------
__global__ void gemm_wmma(const float* __restrict__ A, const __bf16* __restrict__ Bp,
                          const float* __restrict__ bias, const float* __restrict__ addsrc,
                          float* __restrict__ C, int nrt, int K, int N, int do_relu) {
  const int lane = threadIdx.x & 31;
  const int wave = threadIdx.x >> 5;
  const int rt   = blockIdx.x * 8 + wave;     // row tile (wave-uniform guard)
  if (rt >= nrt) return;
  const int ntiles = N >> 4;
  const int nt0    = blockIdx.y * 8;          // this wave's 8 n-tiles
  const int h = lane >> 4;
  const int m = lane & 15;

  v8f acc[8];
#pragma unroll
  for (int t = 0; t < 8; ++t)
    acc[t] = (v8f){0.f, 0.f, 0.f, 0.f, 0.f, 0.f, 0.f, 0.f};

  const float* ar = A + (size_t)(rt * 16 + m) * K;
  const int nkt = K >> 5;
  for (int kt = 0; kt < nkt; ++kt) {
    const float* ap = ar + kt * 32 + 16 * h;
    v16bf a;
#pragma unroll
    for (int i = 0; i < 8; ++i) a[i] = (__bf16)ap[i];
#pragma unroll
    for (int i = 0; i < 8; ++i) a[8 + i] = (__bf16)ap[8 + i];
    const __bf16* bk = Bp + ((size_t)kt * ntiles + nt0) * 512 + (size_t)lane * 16;
#pragma unroll
    for (int t = 0; t < 8; ++t) {
      v16bf b = *(const v16bf*)(bk + (size_t)t * 512);
      acc[t] = __builtin_amdgcn_wmma_f32_16x16x32_bf16(false, a, false, b,
                                                       (short)0, acc[t], false, false);
    }
  }

#pragma unroll
  for (int t = 0; t < 8; ++t) {
    const int col = (nt0 + t) * 16 + m;
    const float bv = bias[col];
#pragma unroll
    for (int v = 0; v < 8; ++v) {
      int r = rt * 16 + 8 * h + v;
      float x = acc[t][v] + bv;
      if (addsrc) x += addsrc[(size_t)r * N + col];
      if (do_relu) x = fmaxf(x, 0.f);
      C[(size_t)r * N + col] = x;
    }
  }
}

// ---------------------------------------------------------------------------
// Edge message: m = relu(x_src[src] + ea*We + be) + GEPS  (recomputed per pass;
// x_src tables are L2-resident, so no [E,H] materialization).
// ---------------------------------------------------------------------------
__device__ __forceinline__ float4 edge_msg(const float* __restrict__ xsrc, int s,
                                           float a, const float* __restrict__ We,
                                           const float* __restrict__ be, int lane) {
  const float4 xs = *(const float4*)(xsrc + (size_t)s * 128 + lane * 4);
  const float4 w  = *(const float4*)(We + lane * 4);
  const float4 bb = *(const float4*)(be + lane * 4);
  float4 m;
  m.x = fmaxf(xs.x + a * w.x + bb.x, 0.f) + GEPS;
  m.y = fmaxf(xs.y + a * w.y + bb.y, 0.f) + GEPS;
  m.z = fmaxf(xs.z + a * w.z + bb.z, 0.f) + GEPS;
  m.w = fmaxf(xs.w + a * w.w + bb.w, 0.f) + GEPS;
  return m;
}

// Pass 1: segment max. All m > 0, so uint-bit atomicMax with 0 init == f32 max
// and reproduces the reference's isfinite(-inf)->0 replacement for empty segs.
__global__ void edge_pass1(const int* __restrict__ src, const int* __restrict__ dst,
                           const float* __restrict__ ea, const float* __restrict__ xsrc,
                           const float* __restrict__ We, const float* __restrict__ be,
                           unsigned* __restrict__ mmax, int E) {
  int e = blockIdx.x * 8 + (threadIdx.x >> 5);
  if (e >= E) return;
  int lane = threadIdx.x & 31;
  int s = src[e], d = dst[e];
  float a = ea[e];
  float4 m = edge_msg(xsrc, s, a, We, be, lane);
  unsigned* p = mmax + (size_t)d * 128 + lane * 4;
  atomicMax(p + 0, __float_as_uint(m.x));
  atomicMax(p + 1, __float_as_uint(m.y));
  atomicMax(p + 2, __float_as_uint(m.z));
  atomicMax(p + 3, __float_as_uint(m.w));
}

// Pass 2: den += e^(m-max), num += m*e^(m-max)   (softmax sum fused: one pass)
__global__ void edge_pass2(const int* __restrict__ src, const int* __restrict__ dst,
                           const float* __restrict__ ea, const float* __restrict__ xsrc,
                           const float* __restrict__ We, const float* __restrict__ be,
                           const float* __restrict__ mmax, float* __restrict__ den,
                           float* __restrict__ num, int E) {
  int e = blockIdx.x * 8 + (threadIdx.x >> 5);
  if (e >= E) return;
  int lane = threadIdx.x & 31;
  int s = src[e], d = dst[e];
  float a = ea[e];
  float4 m = edge_msg(xsrc, s, a, We, be, lane);
  size_t off = (size_t)d * 128 + lane * 4;
  const float4 mx = *(const float4*)(mmax + off);
  float e0 = __expf(m.x - mx.x), e1 = __expf(m.y - mx.y);
  float e2 = __expf(m.z - mx.z), e3 = __expf(m.w - mx.w);
  float* dp = den + off;
  float* np = num + off;
  atomicAdd(dp + 0, e0); atomicAdd(np + 0, m.x * e0);
  atomicAdd(dp + 1, e1); atomicAdd(np + 1, m.y * e1);
  atomicAdd(dp + 2, e2); atomicAdd(np + 2, m.z * e2);
  atomicAdd(dp + 3, e3); atomicAdd(np + 3, m.w * e3);
}

__global__ void zero3(float* __restrict__ a, float* __restrict__ b,
                      float* __restrict__ c, int n) {
  int i = blockIdx.x * 256 + threadIdx.x;
  if (i >= n) return;
  a[i] = 0.f; b[i] = 0.f; c[i] = 0.f;
}

// num = num/(den+1e-16) + x_dst   (aggregated output + root residual)
__global__ void aggout(float* __restrict__ num, const float* __restrict__ den,
                       const float* __restrict__ xdst, int n) {
  int i = blockIdx.x * 256 + threadIdx.x;
  if (i >= n) return;
  num[i] = num[i] / (den[i] + 1e-16f) + xdst[i];
}

__global__ void relu_k(const float* __restrict__ in, float* __restrict__ out, int n) {
  int i = blockIdx.x * 256 + threadIdx.x;
  if (i >= n) return;
  out[i] = fmaxf(in[i], 0.f);
}

// h[i,f] = relu(s[i]*W1[f] + b1[f])   (encoders' rank-1 first layer)
__global__ void rank1_relu(const float* __restrict__ s, const float* __restrict__ w,
                           const float* __restrict__ b, float* __restrict__ outp, int n) {
  int idx = blockIdx.x * 256 + threadIdx.x;
  if (idx >= n * 128) return;
  int i = idx >> 7, f = idx & 127;
  outp[idx] = fmaxf(fmaf(s[i], w[f], b[f]), 0.f);
}

// out[i] = dot(h[i,:], w2) + b2  (pred second layer is a matvec -> wave reduce)
__global__ void pred_final(const float* __restrict__ Hd, const float* __restrict__ w2,
                           const float* __restrict__ b2, float* __restrict__ out, int n) {
  int row = blockIdx.x * 8 + (threadIdx.x >> 5);
  if (row >= n) return;
  int lane = threadIdx.x & 31;
  const float4 h = *(const float4*)(Hd + (size_t)row * 128 + lane * 4);
  const float4 w = *(const float4*)(w2 + lane * 4);
  float s = h.x * w.x + h.y * w.y + h.z * w.z + h.w * w.w;
  for (int o = 16; o > 0; o >>= 1) s += __shfl_xor(s, o, 32);
  if (lane == 0) out[row] = s + b2[0];
}

// ---------------------------------------------------------------------------
extern "C" void kernel_launch(void* const* d_in, const int* in_sizes, int n_in,
                              void* d_out, int out_size, void* d_ws, size_t ws_size,
                              hipStream_t stream) {
  (void)in_sizes; (void)out_size; (void)ws_size;

  const int*   v2c_src = (const int*)d_in[0];
  const int*   v2c_dst = (const int*)d_in[1];
  const int*   c2v_src = (const int*)d_in[2];
  const int*   c2v_dst = (const int*)d_in[3];
  const float* v2c_ea  = (const float*)d_in[4];
  const float* c2v_ea  = (const float*)d_in[5];
  const float* bvec    = (const float*)d_in[8];
  const float* cvec    = (const float*)d_in[9];
  const float* x_start = (const float*)d_in[10];

  // Param leaves in jax pytree order (dicts sorted by key):
  // b_enc{W1,W2,b1,b2}, c_enc{...}, convs[0]{c2v{W1,W2,We,b1,b2,be},v2c{...}},
  // convs[1]{...}, pred{W1,W2,b1,b2}, x_enc{...}
  static const int leaf_off[40] = {
      0, 128, 16512, 16640,                 // b_enc
      16768, 16896, 33280, 33408,           // c_enc
      33536, 66304, 99072, 99200, 99456, 99584,        // L0.c2v
      99712, 132480, 165248, 165376, 165632, 165760,   // L0.v2c
      165888, 198656, 231424, 231552, 231808, 231936,  // L1.c2v
      232064, 264832, 297600, 297728, 297984, 298112,  // L1.v2c
      298240, 314624, 314752, 314880,       // pred
      314881, 315009, 331393, 331521};      // x_enc
  const float* leaf[40];
  if (n_in > 12) {
    for (int i = 0; i < 40 && (11 + i) < n_in; ++i)
      leaf[i] = (const float*)d_in[11 + i];
  } else {
    const float* p = (const float*)d_in[11];
    for (int i = 0; i < 40; ++i) leaf[i] = p + leaf_off[i];
  }

  // Workspace layout: packed bf16 weights first (655360 B), then f32 arrays.
  __bf16* pw = (__bf16*)d_ws;
  float* wf = (float*)((char*)d_ws + 655360);
  float* vals     = wf;                 // [NV,128]
  float* cons     = wf + 12800000;      // [NC,128]
  float* cons_new = wf + 19200000;      // [NC,128]
  float* vals_new = wf + 25600000;      // [NV,128]
  float* mmax     = wf + 38400000;      // [NV,128]
  float* den      = wf + 51200000;      // [NV,128]
  float* num      = wf + 64000000;      // [NV,128]
  float* hidden   = mmax;               // [NV,256] aliases mmax+den (dead then)

  auto pack = [&](const float* W, __bf16* dst, int K, int N) {
    int total = K * N;
    hipLaunchKernelGGL(pack_w, dim3((total + 255) / 256), dim3(256), 0, stream, W, dst, K, N);
  };
  auto gemm = [&](const float* A, const __bf16* Bp, const float* bias,
                  const float* addsrc, float* C, int n, int K, int N, int relu) {
    int nrt = n / 16;
    hipLaunchKernelGGL(gemm_wmma, dim3((nrt + 7) / 8, N / 128), dim3(256), 0, stream,
                       A, Bp, bias, addsrc, C, nrt, K, N, relu);
  };

  // --- Pack weights (bf16 WMMA B layout) ---
  __bf16* p_bW2   = pw + 0;       // b_enc.W2 128x128
  __bf16* p_cW2   = pw + 16384;   // c_enc.W2
  __bf16* p_xW2   = pw + 32768;   // x_enc.W2
  __bf16* p_pW1   = pw + 49152;   // pred.W1
  pack(leaf[1],  p_bW2, 128, 128);
  pack(leaf[5],  p_cW2, 128, 128);
  pack(leaf[37], p_xW2, 128, 128);
  pack(leaf[32], p_pW1, 128, 128);
  const int convbase[4] = {14, 8, 26, 20};  // L0.v2c, L0.c2v, L1.v2c, L1.c2v
  __bf16* p_convW1[4];
  __bf16* p_convW2[4];
  for (int i = 0; i < 4; ++i) {
    p_convW1[i] = pw + 65536  + i * 32768;  // 128x256
    p_convW2[i] = pw + 196608 + i * 32768;  // 256x128
    pack(leaf[convbase[i] + 0], p_convW1[i], 128, 256);
    pack(leaf[convbase[i] + 1], p_convW2[i], 256, 128);
  }

  // --- Encoders ---
  hipLaunchKernelGGL(rank1_relu, dim3((NC * 128 + 255) / 256), dim3(256), 0, stream,
                     bvec, leaf[0], leaf[2], hidden, NC);
  gemm(hidden, p_bW2, leaf[3], nullptr, cons, NC, 128, 128, 0);
  hipLaunchKernelGGL(rank1_relu, dim3((NV * 128 + 255) / 256), dim3(256), 0, stream,
                     x_start, leaf[36], leaf[38], hidden, NV);
  gemm(hidden, p_xW2, leaf[39], nullptr, vals, NV, 128, 128, 0);
  hipLaunchKernelGGL(rank1_relu, dim3((NV * 128 + 255) / 256), dim3(256), 0, stream,
                     cvec, leaf[4], leaf[6], hidden, NV);
  gemm(hidden, p_cW2, leaf[7], vals, vals, NV, 128, 128, 0);

  // --- GENConv: fused 2-pass softmax aggregation + WMMA MLP ---
  auto genconv = [&](int lbase, const __bf16* W1p, const __bf16* W2p,
                     const int* src, const int* dsti, const float* ea,
                     const float* xsrc, const float* xdst, float* outp, int ndst) {
    int nd = ndst * 128;
    hipLaunchKernelGGL(zero3, dim3((nd + 255) / 256), dim3(256), 0, stream,
                       mmax, den, num, nd);
    hipLaunchKernelGGL(edge_pass1, dim3(NEDGE / 8), dim3(256), 0, stream,
                       src, dsti, ea, xsrc, leaf[lbase + 2], leaf[lbase + 5],
                       (unsigned*)mmax, NEDGE);
    hipLaunchKernelGGL(edge_pass2, dim3(NEDGE / 8), dim3(256), 0, stream,
                       src, dsti, ea, xsrc, leaf[lbase + 2], leaf[lbase + 5],
                       mmax, den, num, NEDGE);
    hipLaunchKernelGGL(aggout, dim3((nd + 255) / 256), dim3(256), 0, stream,
                       num, den, xdst, nd);
    gemm(num, W1p, leaf[lbase + 3], nullptr, hidden, ndst, 128, 256, 1);
    gemm(hidden, W2p, leaf[lbase + 4], nullptr, outp, ndst, 256, 128, 0);
  };

  for (int l = 0; l < 2; ++l) {
    int iv = l * 2 + 0, ic = l * 2 + 1;
    genconv(convbase[iv], p_convW1[iv], p_convW2[iv],
            v2c_src, v2c_dst, v2c_ea, vals, cons, cons_new, NC);
    genconv(convbase[ic], p_convW1[ic], p_convW2[ic],
            c2v_src, c2v_dst, c2v_ea, cons_new, vals, vals_new, NV);
    hipLaunchKernelGGL(relu_k, dim3((NC * 128 + 255) / 256), dim3(256), 0, stream,
                       cons_new, cons, NC * 128);
    hipLaunchKernelGGL(relu_k, dim3((NV * 128 + 255) / 256), dim3(256), 0, stream,
                       vals_new, vals, NV * 128);
  }

  // --- Predictor ---
  gemm(vals, p_pW1, leaf[34], nullptr, hidden, NV, 128, 128, 1);
  hipLaunchKernelGGL(pred_final, dim3(NV / 8), dim3(256), 0, stream,
                     hidden, leaf[33], leaf[35], (float*)d_out, NV);
}
